// VanillaRNN_90460601188734
// MI455X (gfx1250) — matrix operand order repaired and checked by reading, not compiled, not run on hardware
//
#include <hip/hip_runtime.h>
#include <hip/hip_bf16.h>
#include <math.h>

// VanillaRNN on MI455X (gfx1250): mixed-precision WMMA (f16 in, f32 accum).
// h_t = tanh(x_t W_ih^T + b_ih + b_hh + h_{t-1} W_hh^T); out = hs W_fc^T + b_fc

typedef __attribute__((ext_vector_type(16))) _Float16 v16h;
typedef __attribute__((ext_vector_type(8)))  float    v8f;
typedef __attribute__((ext_vector_type(4)))  float    f32x4;
typedef __attribute__((ext_vector_type(4)))  _Float16 f16x4;

#define RNN_T 512
#define RNN_B 64
#define RNN_I 512
#define RNN_H 1024
#define RNN_O 512

// ---------------------------------------------------------------------------
// f32 -> f16 conversion, 4 elements per thread, grid-stride
// ---------------------------------------------------------------------------
__global__ __launch_bounds__(256) void cvt_f32_to_f16(const float* __restrict__ in,
                                                      _Float16* __restrict__ out,
                                                      long n4) {
    long i = (long)blockIdx.x * blockDim.x + threadIdx.x;
    long stride = (long)gridDim.x * blockDim.x;
    for (; i < n4; i += stride) {
        f32x4 v = ((const f32x4*)in)[i];
        f16x4 h;
        h.x = (_Float16)v.x; h.y = (_Float16)v.y;
        h.z = (_Float16)v.z; h.w = (_Float16)v.w;
        ((f16x4*)out)[i] = h;
    }
}

// ---------------------------------------------------------------------------
// Fragment loaders (contiguous k order used by this code, same for A and B, wave32)
// A (16x32 f16): lane -> row = lane&15 ; halves 0..15 = K[koff..koff+15],
//                koff = (lane>>4)*16
// B (32x16 f16): lane -> col = lane&15 ; halves 0..15 = K[kb..kb+15],
//                kb = (lane>>4)*16  (weights stored [N][K] row-major = B^T)
// ---------------------------------------------------------------------------
__device__ __forceinline__ v16h load_a_frag(const _Float16* __restrict__ p) {
    v16h f;
    ((f32x4*)&f)[0] = *(const f32x4*)(p);
    ((f32x4*)&f)[1] = *(const f32x4*)(p + 8);
    return f;
}

__device__ __forceinline__ v16h load_b_frag(const _Float16* __restrict__ p) {
    v16h f;
    ((f32x4*)&f)[0] = *(const f32x4*)(p);
    ((f32x4*)&f)[1] = *(const f32x4*)(p + 8);
    return f;
}

// ---------------------------------------------------------------------------
// Generic GEMM: C(MxN,f32) = A(MxK,f16,row-major) x Bw(NxK,f16,row-major)^T
//               + bias1[n] (+ bias2[n] if non-null)
// Block: 256 threads = 8 waves. Block tile 256(M) x 64(N).
// Each wave: 32(M) x 64(N) = 8 accumulators; every B fragment reused by
// 2 A rows, every A fragment reused by 4 B columns -> 1.5 b128 loads / WMMA.
// ---------------------------------------------------------------------------
__global__ __launch_bounds__(256) void gemm_wmma_bias(const _Float16* __restrict__ A,
                                                      const _Float16* __restrict__ Bw,
                                                      const float* __restrict__ bias1,
                                                      const float* __restrict__ bias2,
                                                      float* __restrict__ C,
                                                      int M, int N, int K) {
    const int lane = threadIdx.x & 31;
    const int wave = threadIdx.x >> 5;
    const int block_n = blockIdx.x * 64;
    const int block_m = blockIdx.y * 256;
    const int m0  = block_m + wave * 32;
    const int row = lane & 15;
    const int hi  = lane >> 4;

    v8f acc[2][4];
    #pragma unroll
    for (int i = 0; i < 2; ++i)
        #pragma unroll
        for (int j = 0; j < 4; ++j)
            acc[i][j] = (v8f){};

    const _Float16* Ap0 = A + (size_t)(m0 +  0 + row) * K + hi * 16;
    const _Float16* Ap1 = A + (size_t)(m0 + 16 + row) * K + hi * 16;
    const _Float16* Bp[4] = {
        Bw + (size_t)(block_n +  0 + row) * K + hi * 16,
        Bw + (size_t)(block_n + 16 + row) * K + hi * 16,
        Bw + (size_t)(block_n + 32 + row) * K + hi * 16,
        Bw + (size_t)(block_n + 48 + row) * K + hi * 16,
    };

    for (int kb = 0; kb < K; kb += 32) {
        v16h a0 = load_a_frag(Ap0 + kb);
        v16h a1 = load_a_frag(Ap1 + kb);
        #pragma unroll
        for (int j = 0; j < 4; ++j) {
            v16h b = load_b_frag(Bp[j] + kb);
            acc[0][j] = __builtin_amdgcn_wmma_f32_16x16x32_f16(false, a0, false, b, (short)0, acc[0][j], false, false);
            acc[1][j] = __builtin_amdgcn_wmma_f32_16x16x32_f16(false, a1, false, b, (short)0, acc[1][j], false, false);
        }
    }

    #pragma unroll
    for (int j = 0; j < 4; ++j) {
        const int n = block_n + j * 16 + row;
        float bsum = bias1[n] + (bias2 ? bias2[n] : 0.0f);
        #pragma unroll
        for (int i = 0; i < 2; ++i) {
            #pragma unroll
            for (int r = 0; r < 8; ++r) {
                const int m = m0 + i * 16 + r + 8 * hi;
                C[(size_t)m * N + n] = acc[i][j][r] + bsum;
            }
        }
    }
}

// ---------------------------------------------------------------------------
// One recurrence step: hout = f16(tanh(xp_t + hprev x Whh^T))
// M=64(batch), N=K=H=1024. 128 waves (16 blocks x 8 waves), per-wave 16x32.
// Whh (2MB f16) stays resident in the 192MB L2 across all 512 steps; the
// step is latency-bound, so favor wave-parallelism over operand reuse.
// ---------------------------------------------------------------------------
__global__ __launch_bounds__(256) void rnn_step_wmma(const _Float16* __restrict__ hprev,
                                                     const _Float16* __restrict__ Whh,
                                                     const float* __restrict__ xp_t,
                                                     _Float16* __restrict__ hout) {
    const int lane = threadIdx.x & 31;
    const int wave = threadIdx.x >> 5;
    const int gw = blockIdx.x * 8 + wave;   // 0..127
    const int tm = (gw & 3) * 16;           // M offset: 0,16,32,48
    const int tn = (gw >> 2) * 32;          // N offset: 0..992 step 32
    const int row = lane & 15;
    const int hi  = lane >> 4;

    v8f acc0 = {}, acc1 = {};

    const _Float16* Ap  = hprev + (size_t)(tm + row) * RNN_H + hi * 16;
    const _Float16* Bp0 = Whh + (size_t)(tn +  0 + row) * RNN_H + hi * 16;
    const _Float16* Bp1 = Whh + (size_t)(tn + 16 + row) * RNN_H + hi * 16;

    #pragma unroll 4
    for (int kb = 0; kb < RNN_H; kb += 32) {
        v16h af = load_a_frag(Ap + kb);
        v16h b0 = load_b_frag(Bp0 + kb);
        v16h b1 = load_b_frag(Bp1 + kb);
        acc0 = __builtin_amdgcn_wmma_f32_16x16x32_f16(false, af, false, b0, (short)0, acc0, false, false);
        acc1 = __builtin_amdgcn_wmma_f32_16x16x32_f16(false, af, false, b1, (short)0, acc1, false, false);
    }

    v8f* accs[2] = { &acc0, &acc1 };
    #pragma unroll
    for (int j = 0; j < 2; ++j) {
        const int n = tn + j * 16 + row;
        #pragma unroll
        for (int r = 0; r < 8; ++r) {
            const int m = tm + r + 8 * hi;
            float v = (*accs[j])[r] + xp_t[(size_t)m * RNN_H + n];
            hout[(size_t)m * RNN_H + n] = (_Float16)tanhf(v);
        }
    }
}

// ---------------------------------------------------------------------------
// Host-side orchestration (graph-capture safe: only kernels + memsetAsync)
// ---------------------------------------------------------------------------
extern "C" void kernel_launch(void* const* d_in, const int* in_sizes, int n_in,
                              void* d_out, int out_size, void* d_ws, size_t ws_size,
                              hipStream_t stream) {
    (void)in_sizes; (void)n_in; (void)out_size; (void)ws_size;

    const float* seq  = (const float*)d_in[0];  // [T,B,I]
    const float* W_ih = (const float*)d_in[1];  // [H,I]
    const float* W_hh = (const float*)d_in[2];  // [H,H]
    const float* b_ih = (const float*)d_in[3];  // [H]
    const float* b_hh = (const float*)d_in[4];  // [H]
    const float* W_fc = (const float*)d_in[5];  // [O,H]
    const float* b_fc = (const float*)d_in[6];  // [O]
    float* out = (float*)d_out;                 // [T,B,O] f32

    // Workspace carve-up (256B aligned)
    char* ws = (char*)d_ws;
    size_t off = 0;
    auto carve = [&](size_t bytes) -> void* {
        void* p = ws + off;
        off = (off + bytes + 255) & ~(size_t)255;
        return p;
    };
    const size_t nSeq = (size_t)RNN_T * RNN_B * RNN_I;  // 16.7M
    const size_t nXp  = (size_t)RNN_T * RNN_B * RNN_H;  // 33.5M
    _Float16* seq16 = (_Float16*)carve(nSeq * 2);
    _Float16* wih16 = (_Float16*)carve((size_t)RNN_H * RNN_I * 2);
    _Float16* whh16 = (_Float16*)carve((size_t)RNN_H * RNN_H * 2);
    _Float16* wfc16 = (_Float16*)carve((size_t)RNN_O * RNN_H * 2);
    float*    xp    = (float*)   carve(nXp * 4);
    _Float16* hs    = (_Float16*)carve(nXp * 2);
    _Float16* h0    = (_Float16*)carve((size_t)RNN_B * RNN_H * 2);

    // 1) convert everything to f16 (weights keep [N][K] layout == B^T fragments)
    cvt_f32_to_f16<<<1024, 256, 0, stream>>>(seq,  seq16, (long)(nSeq / 4));
    cvt_f32_to_f16<<<256,  256, 0, stream>>>(W_ih, wih16, (long)((size_t)RNN_H * RNN_I / 4));
    cvt_f32_to_f16<<<256,  256, 0, stream>>>(W_hh, whh16, (long)((size_t)RNN_H * RNN_H / 4));
    cvt_f32_to_f16<<<256,  256, 0, stream>>>(W_fc, wfc16, (long)((size_t)RNN_O * RNN_H / 4));

    // 2) input projection: xp = seq(32768x512) x W_ih^T + b_ih + b_hh
    {
        dim3 grid(RNN_H / 64, (RNN_T * RNN_B) / 256);
        gemm_wmma_bias<<<grid, 256, 0, stream>>>(seq16, wih16, b_ih, b_hh, xp,
                                                 RNN_T * RNN_B, RNN_H, RNN_I);
    }

    // 3) sequential recurrence: 512 captured launches, step t reads hs[t-1]
    hipMemsetAsync(h0, 0, (size_t)RNN_B * RNN_H * 2, stream);
    const size_t stepH = (size_t)RNN_B * RNN_H;
    for (int t = 0; t < RNN_T; ++t) {
        const _Float16* hp = (t == 0) ? h0 : (hs + (size_t)(t - 1) * stepH);
        rnn_step_wmma<<<16, 256, 0, stream>>>(hp, whh16, xp + (size_t)t * stepH,
                                              hs + (size_t)t * stepH);
    }

    // 4) head: out = hs(32768x1024) x W_fc^T + b_fc
    {
        dim3 grid(RNN_O / 64, (RNN_T * RNN_B) / 256);
        gemm_wmma_bias<<<grid, 256, 0, stream>>>(hs, wfc16, b_fc, nullptr, out,
                                                 RNN_T * RNN_B, RNN_O, RNN_H);
    }
}
